// STFNet_78323023610092
// MI455X (gfx1250) — matrix operand order repaired and checked by reading, not compiled, not run on hardware
//
#include <hip/hip_runtime.h>
#include <hip/hip_bf16.h>
#include <stdint.h>

#define TSTEPS 100000
#define SDIM   32
#define ODIM   16
#define ODIM2  256
#define HID    512

typedef float  v2f  __attribute__((ext_vector_type(2)));
typedef float  v8f  __attribute__((ext_vector_type(8)));
typedef __bf16 v16bf __attribute__((ext_vector_type(16)));

union F32Acc { v8f v; float f[8]; };
union BfFrag { v16bf v; unsigned short u[16]; uint4 q[2]; };

__device__ __forceinline__ unsigned short f2bf(float x) {
    union { float f; unsigned u; } c; c.f = x;
    unsigned r = c.u + 0x7FFFu + ((c.u >> 16) & 1u);   // round-to-nearest-even
    return (unsigned short)(r >> 16);
}

__device__ __forceinline__ void ldsfence() {
    asm volatile("s_wait_dscnt 0" ::: "memory");
}

// ---------------------------------------------------------------------------
// Pack a row-major f32 weight matrix [K,N] into bf16 WMMA B-fragment blocks.
// Block (kt,nt) covers K rows [kt*32,kt*32+32), N cols [nt*16,nt*16+16).
// Per-lane layout (wave32): n = lane&15, hi = lane>>4, element i -> K = hi*8+i (i<8), 16+hi*8+(i-8) (i>=8).
// Block id = nt*KT + kt; each lane's 16 bf16 stored contiguously (32B).
// ---------------------------------------------------------------------------
__global__ void pack_b_kernel(const float* __restrict__ W, int K, int N,
                              unsigned short* __restrict__ dst) {
    const int lane = threadIdx.x;
    const int b  = blockIdx.x;
    const int KT = K >> 5;
    const int kt = b % KT;
    const int nt = b / KT;
    const int n  = lane & 15, hi = lane >> 4;
    BfFrag f;
#pragma unroll
    for (int i = 0; i < 16; ++i) {
        int k = kt * 32 + (i < 8 ? hi * 8 + i : 16 + hi * 8 + (i - 8));
        f.u[i] = f2bf(W[(size_t)k * N + nt * 16 + n]);
    }
    uint4* p = (uint4*)(dst + ((size_t)b * 32 + lane) * 16);
    p[0] = f.q[0];
    p[1] = f.q[1];
}

// ---------------------------------------------------------------------------
// MLP: per-wave 16-row tile through all three layers using
// v_wmma_f32_16x16x32_bf16.  Activations staged in LDS (row-major bf16) so
// next layer's A-fragments are two ds_load_b128 per (lane, k-tile).
// ---------------------------------------------------------------------------
__global__ void __launch_bounds__(64) mlp_kernel(
        const float* __restrict__ Z,
        const float* __restrict__ b1, const float* __restrict__ b2,
        const float* __restrict__ b3,
        const unsigned short* __restrict__ W1p,
        const unsigned short* __restrict__ W2p,
        const unsigned short* __restrict__ W3p,
        float* __restrict__ Tm) {
    __shared__ unsigned short h1[2][16 * HID];   // 32 KB
    __shared__ unsigned short h2[2][16 * HID];   // 32 KB
    const int wave = threadIdx.x >> 5;
    const int lane = threadIdx.x & 31;
    const int tile = blockIdx.x * 2 + wave;
    if (tile >= TSTEPS / 16) return;
    const int n = lane & 15, hi = lane >> 4;
    const int rowBase = tile * 16;
    unsigned short* h1w = h1[wave];
    unsigned short* h2w = h2[wave];

    // ---- A fragment of Z tile (16x32 bf16): row m = lane&15,
    //      K runs [hi*8, hi*8+8) and [16+hi*8, 16+hi*8+8)
    BfFrag az;
    {
        const float* zr = Z + (size_t)(rowBase + n) * SDIM;
        float4 z0 = *(const float4*)(zr + hi * 8);
        float4 z1 = *(const float4*)(zr + hi * 8 + 4);
        float4 z2 = *(const float4*)(zr + 16 + hi * 8);
        float4 z3 = *(const float4*)(zr + 16 + hi * 8 + 4);
        az.u[0]  = f2bf(z0.x); az.u[1]  = f2bf(z0.y); az.u[2]  = f2bf(z0.z); az.u[3]  = f2bf(z0.w);
        az.u[4]  = f2bf(z1.x); az.u[5]  = f2bf(z1.y); az.u[6]  = f2bf(z1.z); az.u[7]  = f2bf(z1.w);
        az.u[8]  = f2bf(z2.x); az.u[9]  = f2bf(z2.y); az.u[10] = f2bf(z2.z); az.u[11] = f2bf(z2.w);
        az.u[12] = f2bf(z3.x); az.u[13] = f2bf(z3.y); az.u[14] = f2bf(z3.z); az.u[15] = f2bf(z3.w);
    }

    // ---- layer 1: [16x32] @ [32x512]  (1 k-tile, 32 n-tiles) ----
    for (int nt = 0; nt < HID / 16; ++nt) {
        F32Acc c;
        float bv = b1[nt * 16 + n];
#pragma unroll
        for (int j = 0; j < 8; ++j) c.f[j] = bv;
        BfFrag bw;
        const uint4* bp = (const uint4*)(W1p + ((size_t)nt * 32 + lane) * 16);
        bw.q[0] = bp[0]; bw.q[1] = bp[1];
        c.v = __builtin_amdgcn_wmma_f32_16x16x32_bf16(false, az.v, false, bw.v,
                                                      (short)0, c.v, false, false);
#pragma unroll
        for (int j = 0; j < 8; ++j) {
            float x = c.f[j]; x = x > 0.0f ? x : 0.0f;
            h1w[(hi * 8 + j) * HID + nt * 16 + n] = f2bf(x);
        }
    }
    ldsfence();

    // ---- layer 2: [16x512] @ [512x512]  (16 k-tiles, 32 n-tiles, chunk 4) ----
    for (int ntc = 0; ntc < 8; ++ntc) {
        F32Acc acc[4];
#pragma unroll
        for (int q = 0; q < 4; ++q) {
            float bv = b2[(ntc * 4 + q) * 16 + n];
#pragma unroll
            for (int j = 0; j < 8; ++j) acc[q].f[j] = bv;
        }
        for (int kt = 0; kt < 16; ++kt) {
            BfFrag a;
            const unsigned short* hb = h1w + n * HID + kt * 32 + hi * 8;
            a.q[0] = *(const uint4*)(hb);
            a.q[1] = *(const uint4*)(hb + 16);
#pragma unroll
            for (int q = 0; q < 4; ++q) {
                int blk = (ntc * 4 + q) * 16 + kt;
                BfFrag bw;
                const uint4* bp = (const uint4*)(W2p + ((size_t)blk * 32 + lane) * 16);
                bw.q[0] = bp[0]; bw.q[1] = bp[1];
                acc[q].v = __builtin_amdgcn_wmma_f32_16x16x32_bf16(
                        false, a.v, false, bw.v, (short)0, acc[q].v, false, false);
            }
        }
#pragma unroll
        for (int q = 0; q < 4; ++q)
#pragma unroll
            for (int j = 0; j < 8; ++j) {
                float x = acc[q].f[j]; x = x > 0.0f ? x : 0.0f;
                h2w[(hi * 8 + j) * HID + (ntc * 4 + q) * 16 + n] = f2bf(x);
            }
    }
    ldsfence();

    // ---- layer 3: [16x512] @ [512x256] -> T_mats rows (f32) ----
    for (int nt3 = 0; nt3 < 16; ++nt3) {
        F32Acc c;
        float bv = b3[nt3 * 16 + n];
#pragma unroll
        for (int j = 0; j < 8; ++j) c.f[j] = bv;
        for (int kt = 0; kt < 16; ++kt) {
            BfFrag a;
            const unsigned short* hb = h2w + n * HID + kt * 32 + hi * 8;
            a.q[0] = *(const uint4*)(hb);
            a.q[1] = *(const uint4*)(hb + 16);
            int blk = nt3 * 16 + kt;
            BfFrag bw;
            const uint4* bp = (const uint4*)(W3p + ((size_t)blk * 32 + lane) * 16);
            bw.q[0] = bp[0]; bw.q[1] = bp[1];
            c.v = __builtin_amdgcn_wmma_f32_16x16x32_bf16(false, a.v, false, bw.v,
                                                          (short)0, c.v, false, false);
        }
#pragma unroll
        for (int j = 0; j < 8; ++j)
            Tm[(size_t)(rowBase + hi * 8 + j) * ODIM2 + nt3 * 16 + n] = c.f[j];
    }
}

// ---------------------------------------------------------------------------
// 16x16x16 f32 product via 4 chained v_wmma_f32_16x16x4_f32.
// As, Bs are row-major 16x16 f32 in LDS.  bt=true uses Bs transposed.
// ---------------------------------------------------------------------------
__device__ __forceinline__ v8f mm16(const float* As, const float* Bs, bool bt,
                                    v8f c, int lane) {
    const int m = lane & 15, hi = lane >> 4;
#pragma unroll
    for (int kt = 0; kt < 4; ++kt) {
        const int k0 = kt * 4 + hi * 2;
        v2f a, b;
        a.x = As[m * 16 + k0];
        a.y = As[m * 16 + k0 + 1];
        if (bt) { b.x = Bs[m * 16 + k0];       b.y = Bs[m * 16 + k0 + 1]; }
        else    { b.x = Bs[k0 * 16 + m];       b.y = Bs[(k0 + 1) * 16 + m]; }
        c = __builtin_amdgcn_wmma_f32_16x16x4_f32(false, a, false, b,
                                                  (short)0, c, false, false);
    }
    return c;
}

// ---------------------------------------------------------------------------
// Sequential Kalman scan: one wave, WMMA f32 matmuls + register-resident
// Gauss-Jordan inverse (lane L owns row L&15, aug-cols [hi*16,hi*16+16)).
// ---------------------------------------------------------------------------
__global__ void __launch_bounds__(32) scan_kernel(const float* __restrict__ Tm,
                                                  const float* __restrict__ Y,
                                                  float* __restrict__ out) {
    __shared__ float sT[256], sP[256], sM[256], sPp[256], sSi[256], sK[256];
    __shared__ float sA[16], sApred[16], sW[16], sFac[16], sPivRow[32];
    const int lane = threadIdx.x;
    const int n = lane & 15, hi = lane >> 4;

    // P0 = I, a0 = ones (also first output row)
#pragma unroll
    for (int j = 0; j < 8; ++j)
        sP[(hi * 8 + j) * 16 + n] = ((hi * 8 + j) == n) ? 1.0f : 0.0f;
    if (lane < 16) { sA[lane] = 1.0f; out[lane] = 1.0f; }
    ldsfence();

    for (int t = 1; t < TSTEPS; ++t) {
        // stage T_{t-1} into LDS (coalesced), prefetch next step's matrix
        {
            const float4* src = (const float4*)(Tm + (size_t)(t - 1) * ODIM2);
            float4* dst = (float4*)sT;
            dst[lane]      = src[lane];
            dst[lane + 32] = src[lane + 32];
        }
        if (t < TSTEPS - 1)
            __builtin_prefetch(Tm + (size_t)t * ODIM2 + lane * 8, 0, 1);
        ldsfence();

        F32Acc c;
        // M1 = T * P_prev
#pragma unroll
        for (int j = 0; j < 8; ++j) c.f[j] = 0.0f;
        c.v = mm16(sT, sP, false, c.v, lane);
#pragma unroll
        for (int j = 0; j < 8; ++j) sM[(hi * 8 + j) * 16 + n] = c.f[j];
        ldsfence();

        // P_pred = M1 * T^T + Q  (Q = I)
#pragma unroll
        for (int j = 0; j < 8; ++j) c.f[j] = 0.0f;
        c.v = mm16(sM, sT, true, c.v, lane);
        {
            int dj = n - hi * 8;
            if (dj >= 0 && dj < 8) c.f[dj] += 1.0f;
        }
#pragma unroll
        for (int j = 0; j < 8; ++j) sPp[(hi * 8 + j) * 16 + n] = c.f[j];
        ldsfence();

        // Gauss-Jordan inverse of S = P_pred + H  (H = I), augmented [S | I]
        float g[16];
#pragma unroll
        for (int ci = 0; ci < 16; ++ci) {
            int col = hi * 16 + ci;
            g[ci] = (hi == 0) ? (sPp[n * 16 + col] + ((col == n) ? 1.0f : 0.0f))
                              : (((col - 16) == n) ? 1.0f : 0.0f);
        }
#pragma unroll
        for (int p = 0; p < 16; ++p) {
            if (n == p) {
#pragma unroll
                for (int ci = 0; ci < 16; ++ci) sPivRow[hi * 16 + ci] = g[ci];
            }
            if (hi == 0) sFac[n] = g[p];
            ldsfence();
            float pivinv = 1.0f / sPivRow[p];
            float fac = sFac[n];
            bool isp = (n == p);
#pragma unroll
            for (int ci = 0; ci < 16; ++ci) {
                float rp = sPivRow[hi * 16 + ci] * pivinv;
                g[ci] = isp ? rp : (g[ci] - fac * rp);
            }
            ldsfence();
        }
        if (hi == 1) {
#pragma unroll
            for (int ci = 0; ci < 16; ++ci) sSi[n * 16 + ci] = g[ci];
        }
        ldsfence();

        // K = P_pred * inv(S)
#pragma unroll
        for (int j = 0; j < 8; ++j) c.f[j] = 0.0f;
        c.v = mm16(sPp, sSi, false, c.v, lane);
#pragma unroll
        for (int j = 0; j < 8; ++j) sK[(hi * 8 + j) * 16 + n] = c.f[j];
        ldsfence();

        // P_new = I - K * P_pred   (faithful to reference)
#pragma unroll
        for (int j = 0; j < 8; ++j) c.f[j] = 0.0f;
        c.v = mm16(sK, sPp, false, c.v, lane);
#pragma unroll
        for (int j = 0; j < 8; ++j) {
            int m = hi * 8 + j;
            sP[m * 16 + n] = ((m == n) ? 1.0f : 0.0f) - c.f[j];
        }

        // a update: a_pred = T a ; a_new = a_pred + K (y - a_pred)
        float ap = 0.0f;
        if (lane < 16) {
#pragma unroll
            for (int k2 = 0; k2 < 16; ++k2) ap += sT[lane * 16 + k2] * sA[k2];
            sApred[lane] = ap;
        }
        ldsfence();
        if (lane < 16)
            sW[lane] = Y[(size_t)t * ODIM + lane] - sApred[lane];
        ldsfence();
        if (lane < 16) {
            float an = ap;
#pragma unroll
            for (int k2 = 0; k2 < 16; ++k2) an += sK[lane * 16 + k2] * sW[k2];
            sA[lane] = an;
            out[(size_t)t * ODIM + lane] = an;
        }
        ldsfence();
    }
}

// ---------------------------------------------------------------------------
// Workspace layout (bytes):
//   [0,        32768)   W1 packed bf16 fragments (32 blocks x 1KB)
//   [32768,   557056)   W2 packed (512 blocks)
//   [557056,  819200)   W3 packed (256 blocks)
//   [1048576, ...)      T_mats f32 [100000][256]  (~102.4 MB)
// ---------------------------------------------------------------------------
extern "C" void kernel_launch(void* const* d_in, const int* in_sizes, int n_in,
                              void* d_out, int out_size, void* d_ws, size_t ws_size,
                              hipStream_t stream) {
    const float* Y  = (const float*)d_in[0];
    const float* Z  = (const float*)d_in[1];
    const float* W1 = (const float*)d_in[2];
    const float* b1 = (const float*)d_in[3];
    const float* W2 = (const float*)d_in[4];
    const float* b2 = (const float*)d_in[5];
    const float* W3 = (const float*)d_in[6];
    const float* b3 = (const float*)d_in[7];
    float* out = (float*)d_out;

    char* ws = (char*)d_ws;
    unsigned short* W1p = (unsigned short*)(ws);
    unsigned short* W2p = (unsigned short*)(ws + 32768);
    unsigned short* W3p = (unsigned short*)(ws + 557056);
    float* Tm = (float*)(ws + 1048576);

    pack_b_kernel<<<32,  32, 0, stream>>>(W1, SDIM, HID,   W1p);
    pack_b_kernel<<<512, 32, 0, stream>>>(W2, HID,  HID,   W2p);
    pack_b_kernel<<<256, 32, 0, stream>>>(W3, HID,  ODIM2, W3p);

    mlp_kernel<<<TSTEPS / 32, 64, 0, stream>>>(Z, b1, b2, b3, W1p, W2p, W3p, Tm);

    scan_kernel<<<1, 32, 0, stream>>>(Tm, Y, out);
}
